// RV2BEVFrustumAttn_70815420776723
// MI455X (gfx1250) — matrix operand order repaired and checked by reading, not compiled, not run on hardware
//
#include <hip/hip_runtime.h>
#include <hip/hip_bf16.h>
#include <math.h>

// ---------------------------------------------------------------------------
// Problem constants (B = 1)
// ---------------------------------------------------------------------------
#define NQ     32768      // HRV*WRV = 32*1024
#define NV     16384      // HBEV*WBEV = 128*128
#define NBINS  64
#define CRV    256
#define Y_SIZE   (NQ * 128)        // output y  (B,HRV,WRV,128) row-major
#define DL_SIZE  (NBINS * NQ)      // depth_logits (B,64,HRV,WRV)

typedef __attribute__((ext_vector_type(16))) __bf16       v16bf;
typedef __attribute__((ext_vector_type(8)))  float        v8f;
typedef __attribute__((ext_vector_type(4)))  unsigned int u32x4;
typedef __attribute__((ext_vector_type(4)))  float        f32x4;

union FragU { u32x4 u[2]; v16bf v; };

// ---------------------------------------------------------------------------
// Small device helpers
// ---------------------------------------------------------------------------
__device__ __forceinline__ __bf16 f2bf(float f) {
  union { float f; unsigned u; } a; a.f = f;
  unsigned r = a.u + 0x7fffu + ((a.u >> 16) & 1u);
  union { unsigned short s; __bf16 b; } o;
  o.s = (unsigned short)(r >> 16);
  return o.b;
}

__device__ __forceinline__ float gelu_f(float x) {
  return 0.5f * x * (1.0f + erff(x * 0.70710678118654752f));
}

// BEV sinusoidal positional encoding: pe(c, p), p = y*128+x, C = 128
__device__ __forceinline__ float bev_pe_val(int c, int p) {
  int xx = p & 127, yy = p >> 7;
  float pos; int cc;
  if (c < 64) { pos = (float)xx; cc = c; } else { pos = (float)yy; cc = c - 64; }
  int j = cc >> 1;
  float dv  = __expf((float)j * (-2.0f * 9.210340371976184f / 64.0f)); // exp(-2j*ln(1e4)/64)
  float arg = pos * dv;
  return (cc & 1) ? __cosf(arg) : __sinf(arg);
}

// WMMA fragment loads (ISA 7.12.2 layouts, wave32).
// A 16x32 bf16: lane m = lane&15; VGPR0-3 = K[half*8 .. half*8+7], VGPR4-7 = +16.
__device__ __forceinline__ v16bf load_fragA(const __bf16* row, int kstep, int lane) {
  int half = (lane >> 4) & 1;
  const __bf16* p = row + kstep + half * 8;
  FragU f;
  f.u[0] = *(const u32x4*)(p);
  f.u[1] = *(const u32x4*)(p + 16);
  return f.v;
}
// B 32x16 bf16: lane n = lane&15; VGPR0-3 = K[half*8 .. half*8+7], VGPR4-7 = +16.
__device__ __forceinline__ v16bf load_fragB(const __bf16* row, int kstep, int lane) {
  int half = (lane >> 4) & 1;
  const __bf16* p = row + kstep + half * 8;
  FragU f;
  f.u[0] = *(const u32x4*)(p);
  f.u[1] = *(const u32x4*)(p + 16);
  return f.v;
}

// ---------------------------------------------------------------------------
// Conversion / prep kernels
// ---------------------------------------------------------------------------
__global__ void cvt_rows_bf16(const float* __restrict__ src, __bf16* __restrict__ dst,
                              int total, int cols, int srcld) {
  int i = blockIdx.x * blockDim.x + threadIdx.x;
  if (i >= total) return;
  int r = i / cols, c = i - r * cols;
  dst[i] = f2bf(src[(size_t)r * srcld + c]);
}

// bev (256,128,128) channel-major -> bevT[p*256 + c] bf16
__global__ void transpose_bev_bf16(const float* __restrict__ bev, __bf16* __restrict__ dst) {
  int i = blockIdx.x * blockDim.x + threadIdx.x;
  if (i >= NV * CRV) return;
  int p = i >> 8, c = i & 255;
  dst[i] = f2bf(bev[(size_t)c * NV + p]);
}

// rh2_w (64,128,3,3) -> wtap[t][o][ci] bf16, t = ky*3+kx
__global__ void prep_wtap(const float* __restrict__ w, __bf16* __restrict__ dst) {
  int i = blockIdx.x * blockDim.x + threadIdx.x;
  if (i >= 9 * 64 * 128) return;
  int t = i / 8192, r = i - t * 8192;
  int o = r >> 7, ci = r & 127;
  dst[i] = f2bf(w[(size_t)o * 1152 + ci * 9 + t]);
}

// Lidar ray geometry (ux,uy,uz) per query
__constant__ float c_elev_deg[32] = {
  -30.67f,-29.33f,-28.0f,-26.66f,-25.33f,-24.0f,-22.67f,-21.33f,-20.0f,-18.67f,
  -17.33f,-16.0f,-14.67f,-13.33f,-12.0f,-10.67f,-9.33f,-8.0f,-6.66f,-5.33f,
  -4.0f,-2.67f,-1.33f,0.0f,1.33f,2.67f,4.0f,5.33f,6.67f,8.0f,9.33f,10.67f};

__global__ void geom_kernel(float* ux, float* uy, float* uz) {
  int i = blockIdx.x * blockDim.x + threadIdx.x;
  if (i >= NQ) return;
  int h = i >> 10, w = i & 1023;
  float elev = c_elev_deg[31 - h] * 0.017453292519943295f;
  float az   = -3.14159265358979323846f +
               ((float)w + 0.5f) * (2.0f * 3.14159265358979323846f / 1024.0f);
  float ce = cosf(elev), se = sinf(elev);
  ux[i] = cosf(az) * ce;
  uy[i] = sinf(az) * ce;
  uz[i] = se;
}

// ---------------------------------------------------------------------------
// Generic WMMA GEMM: C[M, NT*16] = A[M,K] * W[NT*16,K]^T (+ flexible epilogue)
// One wave computes a full 16 x (NT*16) strip: the A fragment is loaded ONCE
// per k-step and reused across NT WMMAs (NT accumulators live in VGPRs).
// ---------------------------------------------------------------------------
struct EpArgs {
  const float* bias;                  // [N] or null
  const float* U0; const float* U1; const float* U2;   // [M] rank-update vecs
  const float* V0; const float* V1; const float* V2;   // strided by vstride
  int vstride;
  float*  outF; int ldF;              // row-major f32 store
  __bf16* outB; int ldB;              // row-major bf16 store
  float*  outT; int ldT;              // transposed f32 store at n*ldT + m
  int doGelu;
  int addPE;
};

template <int NT>
__global__ __launch_bounds__(256) void gemm_bf16_wmma(
    const __bf16* __restrict__ A, const __bf16* __restrict__ W,
    int M, int K, EpArgs ep) {
  const int lane = threadIdx.x & 31;
  const int tm   = blockIdx.x * (blockDim.x >> 5) + (threadIdx.x >> 5); // row tile
  if (tm * 16 >= M) return;
  const int l16 = lane & 15;
  const __bf16* arow = A + (size_t)(tm * 16 + l16) * K;
  const __bf16* brow = W + (size_t)l16 * K;          // + t*16*K per tile

  v8f acc[NT];
#pragma unroll
  for (int t = 0; t < NT; ++t) acc[t] = (v8f){0.f,0.f,0.f,0.f,0.f,0.f,0.f,0.f};

  for (int k = 0; k < K; k += 32) {
    v16bf af = load_fragA(arow, k, lane);
#pragma unroll
    for (int t = 0; t < NT; ++t) {
      v16bf bf = load_fragB(brow + (size_t)t * 16 * K, k, lane);
      acc[t] = __builtin_amdgcn_wmma_f32_16x16x32_bf16(false, af, false, bf,
                                                       (short)0, acc[t], false, false);
    }
  }

  const int mb = tm * 16 + ((lane >> 4) << 3);
#pragma unroll
  for (int t = 0; t < NT; ++t) {
    const int n = t * 16 + l16;
    float bias = ep.bias ? ep.bias[n] : 0.0f;
    float v0 = ep.V0 ? ep.V0[(size_t)n * ep.vstride] : 0.0f;
    float v1 = ep.V1 ? ep.V1[(size_t)n * ep.vstride] : 0.0f;
    float v2 = ep.V2 ? ep.V2[(size_t)n * ep.vstride] : 0.0f;
#pragma unroll
    for (int r = 0; r < 8; ++r) {
      int m = mb + r;
      float c = acc[t][r] + bias;
      if (ep.U0) c += ep.U0[m] * v0;
      if (ep.U1) c += ep.U1[m] * v1;
      if (ep.U2) c += ep.U2[m] * v2;
      if (ep.addPE) c += bev_pe_val(n, m);
      if (ep.doGelu) c = gelu_f(c);
      if (ep.outF) ep.outF[(size_t)m * ep.ldF + n] = c;
      if (ep.outB) ep.outB[(size_t)m * ep.ldB + n] = f2bf(c);
      if (ep.outT) ep.outT[(size_t)n * ep.ldT + m] = c;
    }
  }
}

// ---------------------------------------------------------------------------
// 3x3 circular conv as 9 accumulated WMMA GEMMs with row-remapped A.
// One wave computes a 16x64 strip (4 accumulators); A fragment per tap
// loaded once and reused across the 4 output tiles.
// A: h1g [32768,128] bf16, Wt: [9][64][128] bf16, out: h2 [32768,64] f32
// ---------------------------------------------------------------------------
__global__ __launch_bounds__(256) void conv3x3_wmma(
    const __bf16* __restrict__ A, const __bf16* __restrict__ Wt,
    float* __restrict__ outF) {
  const int lane = threadIdx.x & 31;
  const int tm   = blockIdx.x * (blockDim.x >> 5) + (threadIdx.x >> 5);
  if (tm * 16 >= NQ) return;
  const int l16 = lane & 15;
  const int m = tm * 16 + l16;
  const int y = m >> 10, x = m & 1023;

  v8f acc[4];
#pragma unroll
  for (int t = 0; t < 4; ++t) acc[t] = (v8f){0.f,0.f,0.f,0.f,0.f,0.f,0.f,0.f};

#pragma unroll
  for (int tap = 0; tap < 9; ++tap) {
    const int dy = tap / 3 - 1, dx = tap % 3 - 1;
    const int sm = (((y + dy) & 31) << 10) | ((x + dx) & 1023);
    const __bf16* arow = A + (size_t)sm * 128;
    const __bf16* bbase = Wt + (size_t)(tap * 64 + l16) * 128;
    for (int k = 0; k < 128; k += 32) {
      v16bf af = load_fragA(arow, k, lane);
#pragma unroll
      for (int t = 0; t < 4; ++t) {
        v16bf bf = load_fragB(bbase + (size_t)t * 16 * 128, k, lane);
        acc[t] = __builtin_amdgcn_wmma_f32_16x16x32_bf16(false, af, false, bf,
                                                         (short)0, acc[t], false, false);
      }
    }
  }

  const int mb = tm * 16 + ((lane >> 4) << 3);
#pragma unroll
  for (int t = 0; t < 4; ++t) {
    const int n = t * 16 + l16;
#pragma unroll
    for (int r = 0; r < 8; ++r)
      outF[(size_t)(mb + r) * 64 + n] = acc[t][r];
  }
}

// ---------------------------------------------------------------------------
// GroupNorm (two-pass, deterministic)
// ---------------------------------------------------------------------------
__global__ void gn_partial(const float* __restrict__ x, float* __restrict__ part,
                           int M, int C, int cpg) {
  int g = blockIdx.x >> 6;          // 64 blocks per group
  int b = blockIdx.x & 63;
  int mchunk = M >> 6;
  int m0 = b * mchunk;
  int elems = mchunk * cpg;
  float s = 0.f, ss = 0.f;
  for (int t = threadIdx.x; t < elems; t += blockDim.x) {
    int m = m0 + t / cpg;
    int c = g * cpg + (t % cpg);
    float v = x[(size_t)m * C + c];
    s += v; ss += v * v;
  }
  __shared__ float sh[512];
  sh[threadIdx.x] = s; sh[256 + threadIdx.x] = ss;
  __syncthreads();
  for (int o = 128; o > 0; o >>= 1) {
    if ((int)threadIdx.x < o) {
      sh[threadIdx.x] += sh[threadIdx.x + o];
      sh[256 + threadIdx.x] += sh[256 + threadIdx.x + o];
    }
    __syncthreads();
  }
  if (threadIdx.x == 0) {
    part[blockIdx.x * 2 + 0] = sh[0];
    part[blockIdx.x * 2 + 1] = sh[256];
  }
}

__global__ void gn_final(const float* __restrict__ part, float* __restrict__ stats,
                         float invcount) {
  int g = blockIdx.x;
  int t = threadIdx.x;   // 64 threads
  __shared__ float sh[128];
  sh[t]      = part[(g * 64 + t) * 2 + 0];
  sh[64 + t] = part[(g * 64 + t) * 2 + 1];
  __syncthreads();
  for (int o = 32; o > 0; o >>= 1) {
    if (t < o) { sh[t] += sh[t + o]; sh[64 + t] += sh[64 + t + o]; }
    __syncthreads();
  }
  if (t == 0) {
    float mean = sh[0] * invcount;
    float var  = sh[64] * invcount - mean * mean;
    stats[g * 2 + 0] = mean;
    stats[g * 2 + 1] = rsqrtf(var + 1e-5f);
  }
}

__global__ void gn_apply_gelu(const float* __restrict__ x, const float* __restrict__ stats,
                              const float* __restrict__ gamma, const float* __restrict__ beta,
                              __bf16* __restrict__ dst, int M, int C, int cpg) {
  int i = blockIdx.x * blockDim.x + threadIdx.x;
  if (i >= M * C) return;
  int c = i % C;
  int g = c / cpg;
  float v = (x[i] - stats[g * 2]) * stats[g * 2 + 1] * gamma[c] + beta[c];
  dst[i] = f2bf(gelu_f(v));
}

// ---------------------------------------------------------------------------
// Depth softmax: one wave per query; moments, ref points, per-query KL part
// ---------------------------------------------------------------------------
__global__ void depth_stats(const float* __restrict__ dl, const float* __restrict__ ux,
                            const float* __restrict__ uy, const float* __restrict__ uz,
                            const float* __restrict__ l2e,
                            float* mu_n, float* sg_n, float* ref0, float* ref1,
                            float* klpart) {
  int wv   = (blockIdx.x * blockDim.x + threadIdx.x) >> 5;
  int lane = threadIdx.x & 31;
  if (wv >= NQ) return;
  float d0 = dl[(size_t)lane * NQ + wv];
  float d1 = dl[(size_t)(lane + 32) * NQ + wv];
  float mx = fmaxf(d0, d1);
#pragma unroll
  for (int o = 16; o > 0; o >>= 1) mx = fmaxf(mx, __shfl_xor(mx, o, 32));
  float e0 = __expf(d0 - mx), e1 = __expf(d1 - mx);
  float bc0 = ((float)lane + 0.5f) * 0.8f;
  float bc1 = ((float)lane + 32.5f) * 0.8f;
  float s  = e0 + e1;
  float sm = e0 * bc0 + e1 * bc1;
  float s2 = e0 * bc0 * bc0 + e1 * bc1 * bc1;
  float sd = d0 + d1;
#pragma unroll
  for (int o = 16; o > 0; o >>= 1) {
    s  += __shfl_xor(s,  o, 32);
    sm += __shfl_xor(sm, o, 32);
    s2 += __shfl_xor(s2, o, 32);
    sd += __shfl_xor(sd, o, 32);
  }
  if (lane == 0) {
    float inv = 1.0f / s;
    float mu  = sm * inv;
    float var = s2 * inv - mu * mu;
    float sg  = sqrtf(fmaxf(var, 1e-6f));
    float ed  = fminf(mu, 50.8f);                       // RMAX - 0.5*BIN
    mu_n[wv] = fminf(fmaxf(mu * (1.0f / 51.2f), 0.f), 1.f);
    sg_n[wv] = fminf(fmaxf(sg * (1.0f / 51.2f), 0.f), 1.f);
    float xl = ed * ux[wv], yl = ed * uy[wv], zl = ed * uz[wv];
    float pex = xl * l2e[0] + yl * l2e[4] + zl * l2e[8]  + l2e[12];
    float pey = xl * l2e[1] + yl * l2e[5] + zl * l2e[9]  + l2e[13];
    float rx = (pex + 51.2f) * (1.0f / 102.4f);
    float ry = (pey + 51.2f) * (1.0f / 102.4f);
    ref0[wv] = fminf(fmaxf(ry, 0.f), 1.f);
    ref1[wv] = fminf(fmaxf(rx, 0.f), 1.f);
    float lse = mx + __logf(s);
    klpart[wv] = -4.1588830833596715f + lse - sd * (1.0f / 64.0f);  // -log(64)+lse-mean(dl)
  }
}

__global__ void kl_reduce(const float* __restrict__ part, float* __restrict__ out) {
  __shared__ float s[256];
  float a = 0.f;
  for (int i = threadIdx.x; i < NQ; i += 256) a += part[i];
  s[threadIdx.x] = a;
  __syncthreads();
  for (int o = 128; o > 0; o >>= 1) {
    if ((int)threadIdx.x < o) s[threadIdx.x] += s[threadIdx.x + o];
    __syncthreads();
  }
  if (threadIdx.x == 0) out[0] = s[0] * 1e-4f;   // * KLW / B
}

// ---------------------------------------------------------------------------
// Attention-weight softmax over NP=6 within each head
// ---------------------------------------------------------------------------
__global__ void aw_softmax(float* __restrict__ aw) {
  int t = blockIdx.x * blockDim.x + threadIdx.x;   // nq*8 groups
  if (t >= NQ * 8) return;
  float* p = aw + (size_t)t * 6;
  float m = p[0];
#pragma unroll
  for (int i = 1; i < 6; ++i) m = fmaxf(m, p[i]);
  float e[6]; float s = 0.f;
#pragma unroll
  for (int i = 0; i < 6; ++i) { e[i] = __expf(p[i] - m); s += e[i]; }
  float inv = 1.0f / s;
#pragma unroll
  for (int i = 0; i < 6; ++i) p[i] = e[i] * inv;
}

// ---------------------------------------------------------------------------
// MSDA bilinear sampling: thread per (query, head) -> 16 channels
// ---------------------------------------------------------------------------
__global__ void msda_sample(const float* __restrict__ vout, const float* __restrict__ offb,
                            const float* __restrict__ awb, const float* __restrict__ ref0,
                            const float* __restrict__ ref1, __bf16* __restrict__ outb) {
  int t = blockIdx.x * blockDim.x + threadIdx.x;
  if (t >= NQ * 8) return;
  int nq = t >> 3, h = t & 7;
  float r0 = ref0[nq], r1 = ref1[nq];
  float acc[16];
#pragma unroll
  for (int d = 0; d < 16; ++d) acc[d] = 0.f;
  const float* ob = offb + (size_t)nq * 96 + h * 12;
  const float* ab = awb  + (size_t)nq * 48 + h * 6;
  for (int p = 0; p < 6; ++p) {
    float lx = r0 + ob[p * 2 + 0] * (1.0f / 128.0f);
    float ly = r1 + ob[p * 2 + 1] * (1.0f / 128.0f);
    float a  = ab[p];
    float px = lx * 128.0f - 0.5f;
    float py = ly * 128.0f - 0.5f;
    float fx = floorf(px), fy = floorf(py);
    int x0 = (int)fx, y0 = (int)fy;
    float wx = px - fx, wy = py - fy;
    float wgt[4] = {(1.f - wy) * (1.f - wx), (1.f - wy) * wx,
                    wy * (1.f - wx),         wy * wx};
    int xs[4] = {x0, x0 + 1, x0,     x0 + 1};
    int ys[4] = {y0, y0,     y0 + 1, y0 + 1};
#pragma unroll
    for (int q = 0; q < 4; ++q) {
      int xi = xs[q], yi = ys[q];
      if (xi < 0 || xi >= 128 || yi < 0 || yi >= 128) continue;
      const f32x4* src = (const f32x4*)(vout + ((size_t)(yi * 128 + xi) * 128 + h * 16));
      float wv = wgt[q] * a;
#pragma unroll
      for (int d4 = 0; d4 < 4; ++d4) {
        f32x4 v = src[d4];
        acc[d4 * 4 + 0] += wv * v.x;
        acc[d4 * 4 + 1] += wv * v.y;
        acc[d4 * 4 + 2] += wv * v.z;
        acc[d4 * 4 + 3] += wv * v.w;
      }
    }
  }
  __bf16* o = outb + (size_t)nq * 128 + h * 16;
#pragma unroll
  for (int d = 0; d < 16; ++d) o[d] = f2bf(acc[d]);
}

// ---------------------------------------------------------------------------
// Host driver
// ---------------------------------------------------------------------------
extern "C" void kernel_launch(void* const* d_in, const int* in_sizes, int n_in,
                              void* d_out, int out_size, void* d_ws, size_t ws_size,
                              hipStream_t stream) {
  (void)in_sizes; (void)n_in; (void)out_size; (void)ws_size;

  const float* x_rv  = (const float*)d_in[0];
  const float* bev   = (const float*)d_in[1];
  const float* l2e   = (const float*)d_in[2];
  const float* wq_f  = (const float*)d_in[3];  const float* qb   = (const float*)d_in[4];
  const float* wv_f  = (const float*)d_in[5];  const float* vb   = (const float*)d_in[6];
  const float* wo_f  = (const float*)d_in[7];  const float* obi  = (const float*)d_in[8];
  const float* wqd1  = (const float*)d_in[9];  const float* bqd1 = (const float*)d_in[10];
  const float* wqd2  = (const float*)d_in[11]; const float* bqd2 = (const float*)d_in[12];
  const float* wrh1  = (const float*)d_in[13]; const float* brh1 = (const float*)d_in[14];
  const float* g1g   = (const float*)d_in[15]; const float* g1b  = (const float*)d_in[16];
  const float* wrh2  = (const float*)d_in[17];
  const float* g2g   = (const float*)d_in[18]; const float* g2b  = (const float*)d_in[19];
  const float* wrh3  = (const float*)d_in[20]; const float* brh3 = (const float*)d_in[21];
  const float* woff  = (const float*)d_in[22]; const float* boff = (const float*)d_in[23];
  const float* waw   = (const float*)d_in[24]; const float* baw  = (const float*)d_in[25];
  const float* wvp   = (const float*)d_in[26]; const float* bvp  = (const float*)d_in[27];
  const float* wop   = (const float*)d_in[28]; const float* bop  = (const float*)d_in[29];

  float* outY  = (float*)d_out;                 // [nq,128]
  float* outDL = (float*)d_out + Y_SIZE;        // [64,nq]
  float* outKL = (float*)d_out + Y_SIZE + DL_SIZE;

  // ---- workspace bump allocator --------------------------------------------
  char* wptr = (char*)d_ws;
  auto alloc = [&](size_t bytes) -> char* {
    char* p = wptr;
    wptr += (bytes + 255) & ~(size_t)255;
    return p;
  };
  __bf16* xbf        = (__bf16*)alloc((size_t)NQ * 256 * 2);
  __bf16* q0_bf      = (__bf16*)alloc((size_t)NQ * 128 * 2);
  float*  h1f        = (float*) alloc((size_t)NQ * 128 * 4);   // reused as offbuf
  __bf16* h1g_bf     = (__bf16*)alloc((size_t)NQ * 128 * 2);
  float*  h2f        = (float*) alloc((size_t)NQ * 64 * 4);    // reused as awbuf
  __bf16* h2g_bf     = (__bf16*)alloc((size_t)NQ * 64 * 2);
  __bf16* bevT_bf    = (__bf16*)alloc((size_t)NV * 256 * 2);
  __bf16* vmap_bf    = (__bf16*)alloc((size_t)NV * 128 * 2);
  float*  vout       = (float*) alloc((size_t)NV * 128 * 4);
  __bf16* q1g_bf     = (__bf16*)alloc((size_t)NQ * 128 * 2);   // reused as outmsda
  __bf16* query_bf   = (__bf16*)alloc((size_t)NQ * 128 * 2);
  float*  uxb        = (float*) alloc(NQ * 4);
  float*  uyb        = (float*) alloc(NQ * 4);
  float*  uzb        = (float*) alloc(NQ * 4);
  float*  mu_n       = (float*) alloc(NQ * 4);
  float*  sg_n       = (float*) alloc(NQ * 4);
  float*  ref0       = (float*) alloc(NQ * 4);
  float*  ref1       = (float*) alloc(NQ * 4);
  float*  klpart     = (float*) alloc(NQ * 4);
  float*  gnpart     = (float*) alloc(8 * 64 * 2 * 4);
  float*  gnstats    = (float*) alloc(8 * 2 * 4);
  // bf16 weights
  __bf16* bwq   = (__bf16*)alloc(128 * 256 * 2);
  __bf16* bwv   = (__bf16*)alloc(128 * 256 * 2);
  __bf16* bwpo  = (__bf16*)alloc(128 * 128 * 2);
  __bf16* bwqd1 = (__bf16*)alloc(128 * 128 * 2);
  __bf16* bwqd2 = (__bf16*)alloc(128 * 128 * 2);
  __bf16* bwrh1 = (__bf16*)alloc(128 * 256 * 2);
  __bf16* bwrh3 = (__bf16*)alloc(64 * 64 * 2);
  __bf16* bwoff = (__bf16*)alloc(96 * 128 * 2);
  __bf16* bwaw  = (__bf16*)alloc(48 * 128 * 2);
  __bf16* bwvp  = (__bf16*)alloc(128 * 128 * 2);
  __bf16* bwop  = (__bf16*)alloc(128 * 128 * 2);
  __bf16* bwtap = (__bf16*)alloc(9 * 64 * 128 * 2);
  // lifetime overlays
  float*  offbuf    = h1f;          // [nq,96]  (h1f dead after gn1 apply)
  float*  awbuf     = h2f;          // [nq,48]  (h2f dead after gn2 apply)
  __bf16* outmsda   = q1g_bf;       // (q1g dead after qd2)
  __bf16* ymsda_bf  = q0_bf;        // (q0 dead after qd1)

  auto blocks = [](long n) { return dim3((unsigned)((n + 255) / 256)); };
  auto gemm = [&](const __bf16* A, const __bf16* W, int M, int N, int K, EpArgs ep) {
    int rowtiles = M / 16;
    dim3 g((rowtiles + 7) / 8), b(256);
    switch (N >> 4) {
      case 8: gemm_bf16_wmma<8><<<g, b, 0, stream>>>(A, W, M, K, ep); break;
      case 6: gemm_bf16_wmma<6><<<g, b, 0, stream>>>(A, W, M, K, ep); break;
      case 4: gemm_bf16_wmma<4><<<g, b, 0, stream>>>(A, W, M, K, ep); break;
      case 3: gemm_bf16_wmma<3><<<g, b, 0, stream>>>(A, W, M, K, ep); break;
      default: break;
    }
  };

  // ---- 0. weight / input conversions ---------------------------------------
  cvt_rows_bf16<<<blocks(128 * 256), 256, 0, stream>>>(wq_f,  bwq,   128 * 256, 256, 256);
  cvt_rows_bf16<<<blocks(128 * 256), 256, 0, stream>>>(wv_f,  bwv,   128 * 256, 256, 256);
  cvt_rows_bf16<<<blocks(128 * 128), 256, 0, stream>>>(wo_f,  bwpo,  128 * 128, 128, 128);
  cvt_rows_bf16<<<blocks(128 * 128), 256, 0, stream>>>(wqd1,  bwqd1, 128 * 128, 128, 130);
  cvt_rows_bf16<<<blocks(128 * 128), 256, 0, stream>>>(wqd2,  bwqd2, 128 * 128, 128, 128);
  cvt_rows_bf16<<<blocks(128 * 256), 256, 0, stream>>>(wrh1,  bwrh1, 128 * 256, 256, 259);
  cvt_rows_bf16<<<blocks(64 * 64),   256, 0, stream>>>(wrh3,  bwrh3, 64 * 64,   64,  64);
  cvt_rows_bf16<<<blocks(96 * 128),  256, 0, stream>>>(woff,  bwoff, 96 * 128,  128, 128);
  cvt_rows_bf16<<<blocks(48 * 128),  256, 0, stream>>>(waw,   bwaw,  48 * 128,  128, 128);
  cvt_rows_bf16<<<blocks(128 * 128), 256, 0, stream>>>(wvp,   bwvp,  128 * 128, 128, 128);
  cvt_rows_bf16<<<blocks(128 * 128), 256, 0, stream>>>(wop,   bwop,  128 * 128, 128, 128);
  prep_wtap<<<blocks(9 * 64 * 128), 256, 0, stream>>>(wrh2, bwtap);
  cvt_rows_bf16<<<blocks((long)NQ * 256), 256, 0, stream>>>(x_rv, xbf, NQ * 256, 256, 256);
  transpose_bev_bf16<<<blocks((long)NV * 256), 256, 0, stream>>>(bev, bevT_bf);
  geom_kernel<<<blocks(NQ), 256, 0, stream>>>(uxb, uyb, uzb);

  // ---- 1. Q0 = proj_q(x) ----------------------------------------------------
  { EpArgs ep{}; ep.bias = qb; ep.outB = q0_bf; ep.ldB = 128;
    gemm(xbf, bwq, NQ, 128, 256, ep); }

  // ---- 2. range head: rh1 (K=256 GEMM + rank-3 geometry tail) --------------
  { EpArgs ep{}; ep.bias = brh1;
    ep.U0 = uxb; ep.U1 = uyb; ep.U2 = uzb;
    ep.V0 = wrh1 + 256; ep.V1 = wrh1 + 257; ep.V2 = wrh1 + 258; ep.vstride = 259;
    ep.outF = h1f; ep.ldF = 128;
    gemm(xbf, bwrh1, NQ, 128, 256, ep); }
  gn_partial<<<dim3(8 * 64), 256, 0, stream>>>(h1f, gnpart, NQ, 128, 16);
  gn_final<<<dim3(8), 64, 0, stream>>>(gnpart, gnstats, 1.0f / (32768.0f * 16.0f));
  gn_apply_gelu<<<blocks((long)NQ * 128), 256, 0, stream>>>(h1f, gnstats, g1g, g1b,
                                                            h1g_bf, NQ, 128, 16);
  conv3x3_wmma<<<dim3(2048 / 8), dim3(256), 0, stream>>>(h1g_bf, bwtap, h2f);
  gn_partial<<<dim3(8 * 64), 256, 0, stream>>>(h2f, gnpart, NQ, 64, 8);
  gn_final<<<dim3(8), 64, 0, stream>>>(gnpart, gnstats, 1.0f / (32768.0f * 8.0f));
  gn_apply_gelu<<<blocks((long)NQ * 64), 256, 0, stream>>>(h2f, gnstats, g2g, g2b,
                                                           h2g_bf, NQ, 64, 8);
  { EpArgs ep{}; ep.bias = brh3; ep.outT = outDL; ep.ldT = NQ;   // depth_logits [64,nq]
    gemm(h2g_bf, bwrh3, NQ, 64, 64, ep); }

  // ---- 3. depth softmax moments, ref points, KL ----------------------------
  depth_stats<<<blocks((long)NQ * 32), 256, 0, stream>>>(outDL, uxb, uyb, uzb, l2e,
                                                         mu_n, sg_n, ref0, ref1, klpart);
  kl_reduce<<<dim3(1), 256, 0, stream>>>(klpart, outKL);

  // ---- 4. BEV value path: proj_v (+PE) then vp -----------------------------
  { EpArgs ep{}; ep.bias = vb; ep.addPE = 1; ep.outB = vmap_bf; ep.ldB = 128;
    gemm(bevT_bf, bwv, NV, 128, 256, ep); }
  { EpArgs ep{}; ep.bias = bvp; ep.outF = vout; ep.ldF = 128;
    gemm(vmap_bf, bwvp, NV, 128, 128, ep); }

  // ---- 5. query head: qd1 (rank-2 mu/sigma tail) + gelu, qd2 ---------------
  { EpArgs ep{}; ep.bias = bqd1;
    ep.U0 = mu_n; ep.U1 = sg_n;
    ep.V0 = wqd1 + 128; ep.V1 = wqd1 + 129; ep.vstride = 130;
    ep.doGelu = 1; ep.outB = q1g_bf; ep.ldB = 128;
    gemm(q0_bf, bwqd1, NQ, 128, 128, ep); }
  { EpArgs ep{}; ep.bias = bqd2; ep.outB = query_bf; ep.ldB = 128;
    gemm(q1g_bf, bwqd2, NQ, 128, 128, ep); }

  // ---- 6. offsets / attention weights --------------------------------------
  { EpArgs ep{}; ep.bias = boff; ep.outF = offbuf; ep.ldF = 96;
    gemm(query_bf, bwoff, NQ, 96, 128, ep); }
  { EpArgs ep{}; ep.bias = baw; ep.outF = awbuf; ep.ldF = 48;
    gemm(query_bf, bwaw, NQ, 48, 128, ep); }
  aw_softmax<<<blocks((long)NQ * 8), 256, 0, stream>>>(awbuf);

  // ---- 7. MSDA bilinear sampling -------------------------------------------
  msda_sample<<<blocks((long)NQ * 8), 256, 0, stream>>>(vout, offbuf, awbuf,
                                                        ref0, ref1, outmsda);

  // ---- 8. output projections: op then proj_o -> d_out ----------------------
  { EpArgs ep{}; ep.bias = bop; ep.outB = ymsda_bf; ep.ldB = 128;
    gemm(outmsda, bwop, NQ, 128, 128, ep); }
  { EpArgs ep{}; ep.bias = obi; ep.outF = outY; ep.ldF = 128;
    gemm(ymsda_bf, bwpo, NQ, 128, 128, ep); }
}
